// CNN_LSTM_45363444580399
// MI455X (gfx1250) — matrix operand order repaired and checked by reading, compiled, not run on hardware
//
#include <hip/hip_runtime.h>
#include <cstdint>
#include <cstddef>

// ---------------------------------------------------------------- constants
#define MAXN  25600        // 128 docs * 200 max len
#define DPAD  1088         // 1068 padded up to a multiple of 32
#define DIN   1068
#define FHID  768          // 256*3 conv channels == HIDDEN
#define G4    3072         // 4*HIDDEN gates
#define SHIFT 9            // SPACE/2 + 1

typedef unsigned short u16;
typedef __attribute__((ext_vector_type(16))) __bf16 v16bf;
typedef __attribute__((ext_vector_type(8)))  float  v8f;

union FragB {
  v16bf v;
  u16   u[16];
  uint4 q[2];
};

// ---------------------------------------------------------------- helpers
__device__ __forceinline__ u16 f2bf(float f) {
  unsigned int b = __float_as_uint(f);
  b += 0x7FFFu + ((b >> 16) & 1u);          // round-to-nearest-even
  return (u16)(b >> 16);
}
__device__ __forceinline__ float bf2f(u16 h) {
  return __uint_as_float(((unsigned int)h) << 16);
}
__device__ __forceinline__ float sigm(float x) { return 1.f / (1.f + __expf(-x)); }

// A fragment (16x32 bf16): lane m = lane&15, half = lane>>4.
// K(e) = (e<8 ? half*8+e : 16+half*8+(e-8))  -> two contiguous 16B runs.
__device__ __forceinline__ void load_a(FragB& f, const u16* p, int half) {
  f.q[0] = *reinterpret_cast<const uint4*>(p + half * 8);
  f.q[1] = *reinterpret_cast<const uint4*>(p + 16 + half * 8);
}
// B fragment (32x16 bf16), built from row-major W[n][k]: lane n = lane&15,
// K(e) = (e<8 ? half*8+e : 16+half*8+(e-8))  -> two contiguous 16B runs.
__device__ __forceinline__ void load_b(FragB& f, const u16* p, int half) {
  f.q[0] = *reinterpret_cast<const uint4*>(p + half * 8);
  f.q[1] = *reinterpret_cast<const uint4*>(p + 16 + half * 8);
}
__device__ __forceinline__ v8f wmma_bf(const FragB& a, const FragB& b, v8f c) {
  return __builtin_amdgcn_wmma_f32_16x16x32_bf16(false, a.v, false, b.v,
                                                 (short)0, c, false, false);
}

// CDNA5 async Global->LDS DMA (ASYNCcnt). lds_off is the byte offset inside
// this kernel's (single) LDS object, which is allocated at wave-LDS base 0.
__device__ __forceinline__ void async_g2l_b128(unsigned lds_off, const void* gsrc) {
  asm volatile("global_load_async_to_lds_b128 %0, %1, off"
               :: "v"(lds_off), "v"((unsigned long)gsrc) : "memory");
}
__device__ __forceinline__ void wait_async0() {
  asm volatile("s_wait_asynccnt 0" ::: "memory");
}

// ---------------------------------------------------------------- tiny kernels
__global__ void k_offsets(const int* lens, int* offs) {
  if (threadIdx.x == 0) {
    int s = 0;
    for (int d = 0; d < 128; ++d) { offs[d] = s; s += lens[d]; }
    offs[128] = s;
  }
}

__global__ void k_posw(const int* offs, int* posw) {
  int d = blockIdx.x;
  int s = offs[d], e = offs[d + 1];
  for (int i = threadIdx.x; s + i < e; i += blockDim.x) posw[s + i] = i;
}

// x_doc[n][j] = bf16( x[n][j] + PE(posw[n], j) ), zero-padded to DPAD cols
__global__ void k_embed(const float* __restrict__ x, const int* __restrict__ posw,
                        u16* __restrict__ xdoc, int N) {
  long id = (long)blockIdx.x * blockDim.x + threadIdx.x;
  long total = (long)N * DPAD;
  if (id >= total) return;
  int n = (int)(id / DPAD), j = (int)(id % DPAD);
  float v = 0.f;
  if (j < DIN) {
    float pos = (float)posw[n];
    int i = j >> 1;
    float ang = pos * __expf(-(2.0f * (float)i / (float)DIN) * 9.21034037198f); // ln(1e4)
    float pe = (j & 1) ? __cosf(ang) : __sinf(ang);
    v = x[(size_t)n * DIN + j] + pe;
  }
  xdoc[(size_t)n * DPAD + j] = f2bf(v);
}

__global__ void k_fill_u16(u16* dst, u16 val, long count) {
  long id = (long)blockIdx.x * blockDim.x + threadIdx.x;
  if (id < count) dst[id] = val;
}

__global__ void k_pack_bf(const float* src, u16* dst, long count) {
  long id = (long)blockIdx.x * blockDim.x + threadIdx.x;
  if (id < count) dst[id] = f2bf(src[id]);
}

__global__ void k_pack_bias(const float* b1, const float* b2, float* dst, int count) {
  int id = blockIdx.x * blockDim.x + threadIdx.x;
  if (id < count) dst[id] = b1[id] + b2[id];
}

// conv weights w[oc][c][t] (f32) -> wp[t][oc][DPAD] (bf16, zero pad c>=1068)
__global__ void k_pack_conv(const float* __restrict__ w, u16* __restrict__ wp, int taps) {
  long id = (long)blockIdx.x * blockDim.x + threadIdx.x;
  long total = (long)taps * 256 * DPAD;
  if (id >= total) return;
  int t  = (int)(id / (256 * DPAD));
  int r  = (int)(id % (256 * DPAD));
  int oc = r / DPAD, c = r % DPAD;
  float v = (c < DIN) ? w[((size_t)oc * DIN + c) * taps + t] : 0.f;
  wp[((size_t)t * 256 + oc) * DPAD + c] = f2bf(v);
}

// ---------------------------------------------------------------- conv (implicit GEMM)
// block = 256 thr (8 waves). Wave w computes cols [w*32, w*32+32) of a 16-row strip.
__global__ void k_conv(const u16* __restrict__ xdoc, const u16* __restrict__ wp,
                       const float* __restrict__ bias, u16* __restrict__ ycat,
                       int taps, int pad) {
  const int lane = threadIdx.x & 31, wv = threadIdx.x >> 5;
  const int half = lane >> 4, l15 = lane & 15;
  const int rowbase = blockIdx.x * 16;
  const int c0 = wv * 32;
  v8f acc0 = {}, acc1 = {};
  for (int t = 0; t < taps; ++t) {
    const u16* arow = xdoc + (size_t)(rowbase + l15 + t - pad) * DPAD;
    const u16* w0 = wp + ((size_t)t * 256 + c0 + l15) * DPAD;
    const u16* w1 = wp + ((size_t)t * 256 + c0 + 16 + l15) * DPAD;
    for (int kk = 0; kk < DPAD; kk += 32) {
      FragB a, b0, b1;
      load_a(a, arow + kk, half);
      load_b(b0, w0 + kk, half);
      load_b(b1, w1 + kk, half);
      acc0 = wmma_bf(a, b0, acc0);
      acc1 = wmma_bf(a, b1, acc1);
    }
  }
#pragma unroll
  for (int r = 0; r < 8; ++r) {
    int row = rowbase + r + 8 * half;
    float v0 = acc0[r] + bias[c0 + l15];
    float v1 = acc1[r] + bias[c0 + 16 + l15];
    v0 = v0 >= 0.f ? v0 : 0.01f * v0;        // LeakyReLU
    v1 = v1 >= 0.f ? v1 : 0.01f * v1;
    ycat[(size_t)row * FHID + c0 + l15]      = f2bf(v0);
    ycat[(size_t)row * FHID + c0 + 16 + l15] = f2bf(v1);
  }
}

// maxpool (stride 1, same, -inf halo rows) -> pooled bf16; zero for pad rows
__global__ void k_pool(const u16* __restrict__ ycat, u16* __restrict__ pooled,
                       int N, int Npad) {
  long id = (long)blockIdx.x * blockDim.x + threadIdx.x;
  long total = (long)Npad * FHID;
  if (id >= total) return;
  int n = (int)(id / FHID), f = (int)(id % FHID);
  if (n >= N) { pooled[id] = 0; return; }
  int p = (f < 256) ? 1 : ((f < 512) ? 2 : 3);
  float m = -__builtin_inff();
  for (int t = -p; t <= p; ++t) m = fmaxf(m, bf2f(ycat[(size_t)(n + t) * FHID + f]));
  pooled[id] = f2bf(m);
}

// ---------------------------------------------------------------- gate GEMM
// gout[row][0..3072) = X[row] @ Wx^T (+ H[row+shift] @ Wh^T) + bsum
// block = 256 thr (8 waves); wave handles 2 N-tiles; grid = (Mtiles, 3072/256).
// The 16x768 bf16 A strips (X and shifted H) are staged once per block into
// LDS with CDNA5 async Global->LDS DMA and shared by all 8 waves.
__global__ void k_gemm_gates(const u16* __restrict__ X, const u16* __restrict__ H,
                             int shift, const u16* __restrict__ Wx,
                             const u16* __restrict__ Wh, const float* __restrict__ bsum,
                             float* __restrict__ gout, int N) {
  __shared__ u16 stage[32 * FHID];             // [0,24KB) = X strip, [24KB,48KB) = H strip
  u16* ash = stage;
  u16* hsh = stage + 16 * FHID;
  const int tid = threadIdx.x;
  const int lane = tid & 31, wv = tid >> 5;
  const int half = lane >> 4, l15 = lane & 15;
  const int rowbase = blockIdx.x * 16;
  const int n0 = blockIdx.y * 256 + wv * 32;

  // ---- stage X strip: 16 contiguous rows = 24576B = 1536 x b128 chunks
  {
    const u16* src = X + (size_t)rowbase * FHID;
#pragma unroll
    for (int it = 0; it < 6; ++it) {
      int idx = tid + it * 256;
      async_g2l_b128((unsigned)(idx * 16), src + (size_t)idx * 8);
    }
  }
  // ---- stage H strip with row-shift + boundary zeroing (uniform H test)
  if (H) {
#pragma unroll
    for (int it = 0; it < 3; ++it) {
      int idx = tid + it * 256;                // 768 x b128 chunks (48 per row)
      int m = idx / 48, kq = idx % 48;
      int hrow = rowbase + m + shift;
      if (hrow >= 0 && hrow < N) {
        async_g2l_b128((unsigned)(16 * FHID * 2 + idx * 16),
                       H + (size_t)hrow * FHID + (size_t)kq * 8);
      } else {
        *reinterpret_cast<uint4*>(hsh + (size_t)idx * 8) = make_uint4(0, 0, 0, 0);
      }
    }
  }
  wait_async0();                               // this wave's DMA done
  __syncthreads();                             // whole block's DMA visible

  v8f acc0 = {}, acc1 = {};
  {
    const u16* xr = ash + (size_t)l15 * FHID;
    const u16* w0 = Wx + (size_t)(n0 + l15) * FHID;
    const u16* w1 = Wx + (size_t)(n0 + 16 + l15) * FHID;
    for (int kk = 0; kk < FHID; kk += 32) {
      FragB a, b0, b1;
      load_a(a, xr + kk, half);
      load_b(b0, w0 + kk, half);
      load_b(b1, w1 + kk, half);
      acc0 = wmma_bf(a, b0, acc0);
      acc1 = wmma_bf(a, b1, acc1);
    }
  }
  if (H) {   // uniform branch: EXEC stays all-ones around the WMMAs
    const u16* hr = hsh + (size_t)l15 * FHID;
    const u16* w0 = Wh + (size_t)(n0 + l15) * FHID;
    const u16* w1 = Wh + (size_t)(n0 + 16 + l15) * FHID;
    for (int kk = 0; kk < FHID; kk += 32) {
      FragB a, b0, b1;
      load_a(a, hr + kk, half);
      load_b(b0, w0 + kk, half);
      load_b(b1, w1 + kk, half);
      acc0 = wmma_bf(a, b0, acc0);
      acc1 = wmma_bf(a, b1, acc1);
    }
  }
#pragma unroll
  for (int r = 0; r < 8; ++r) {
    int row = rowbase + r + 8 * half;
    gout[(size_t)row * G4 + n0 + l15]      = acc0[r] + bsum[n0 + l15];
    gout[(size_t)row * G4 + n0 + 16 + l15] = acc1[r] + bsum[n0 + 16 + l15];
  }
}

// ---------------------------------------------------------------- persistent scan
// one workgroup, 32 waves; h (bf16) + c (f32) + activated gates in 264KB LDS
// (only possible on CDNA5's 320KB-per-WGP LDS).
__global__ void __launch_bounds__(1024)
k_scan(const float* __restrict__ gates, const u16* __restrict__ Whh,
       u16* __restrict__ h_all, float* __restrict__ c_all, int N, int nChunks) {
  extern __shared__ char smem[];
  u16*   h_sh = (u16*)smem;                                     // 16*768*2 = 24KB
  float* c_sh = (float*)(smem + 16 * FHID * 2);                 // 48KB
  float* gact = (float*)(smem + 16 * FHID * 2 + 16 * FHID * 4); // 16*3072*4 = 192KB
  const int tid = threadIdx.x;
  const int lane = tid & 31, wv = tid >> 5;                     // wv: 0..31
  const int half = lane >> 4, l15 = lane & 15;
  for (int i = tid; i < 16 * FHID; i += 1024) { h_sh[i] = 0; c_sh[i] = 0.f; }
  __syncthreads();

  for (int q = 0; q < nChunks; ++q) {
    // init accumulators with the precomputed x-side gates (+bias)
    v8f acc[6];
#pragma unroll
    for (int t = 0; t < 6; ++t) {
      int n = wv * 96 + t * 16 + l15;
#pragma unroll
      for (int r = 0; r < 8; ++r)
        acc[t][r] = gates[(size_t)(q * 16 + r + 8 * half) * G4 + n];
    }
    // h @ Whh^T : K=768, Whh stays L2-resident (4.7MB bf16 << 192MB L2)
    for (int kk = 0; kk < FHID; kk += 32) {
      FragB a;
      load_a(a, h_sh + (size_t)l15 * FHID + kk, half);
      __builtin_prefetch(Whh + (size_t)(wv * 96 + l15) * FHID + kk + 32, 0, 3);
#pragma unroll
      for (int t = 0; t < 6; ++t) {
        FragB b;
        load_b(b, Whh + (size_t)(wv * 96 + t * 16 + l15) * FHID + kk, half);
        acc[t] = wmma_bf(a, b, acc[t]);
      }
    }
    // activate (sigmoid for i/f/o, tanh for g) into LDS
#pragma unroll
    for (int t = 0; t < 6; ++t) {
      int n = wv * 96 + t * 16 + l15;
      bool isg = (n >= 1536) && (n < 2304);
#pragma unroll
      for (int r = 0; r < 8; ++r) {
        float v = acc[t][r];
        gact[(r + 8 * half) * G4 + n] = isg ? tanhf(v) : sigm(v);
      }
    }
    __syncthreads();
    // combine gates -> new h, c (16*768 = 12 elems/thread)
#pragma unroll
    for (int e = 0; e < 12; ++e) {
      int id = tid + e * 1024;           // id == m*768 + j
      int m = id / FHID, j = id % FHID;
      float iv = gact[m * G4 + j];
      float fv = gact[m * G4 + FHID + j];
      float gv = gact[m * G4 + 2 * FHID + j];
      float ov = gact[m * G4 + 3 * FHID + j];
      float c2 = fv * c_sh[id] + iv * gv;
      float hv = ov * tanhf(c2);
      c_sh[id] = c2;
      h_sh[id] = f2bf(hv);
      int row = q * 16 + m;
      if (row < N) {
        h_all[(size_t)row * FHID + j] = f2bf(hv);
        c_all[(size_t)row * FHID + j] = c2;
      }
    }
    __syncthreads();
  }
}

// final cell activation: out[n][obase+j] = sig(o)*tanh(sig(f)*c_prev + sig(i)*tanh(g))
__global__ void k_cellact(const float* __restrict__ gates, const float* __restrict__ c_all,
                          int shift, float* __restrict__ out, int obase, int N) {
  long id = (long)blockIdx.x * blockDim.x + threadIdx.x;
  long total = (long)N * FHID;
  if (id >= total) return;
  int n = (int)(id / FHID), j = (int)(id % FHID);
  const float* g = gates + (size_t)n * G4;
  float iv = g[j], fv = g[FHID + j], gv = g[2 * FHID + j], ov = g[3 * FHID + j];
  int m = n + shift;
  float cp = (m >= 0 && m < N) ? c_all[(size_t)m * FHID + j] : 0.f;
  float c2 = sigm(fv) * cp + sigm(iv) * tanhf(gv);
  out[(size_t)n * (2 * FHID) + obase + j] = sigm(ov) * tanhf(c2);
}

// ---------------------------------------------------------------- launcher
extern "C" void kernel_launch(void* const* d_in, const int* in_sizes, int n_in,
                              void* d_out, int out_size, void* d_ws, size_t ws_size,
                              hipStream_t stream) {
  (void)in_sizes; (void)n_in; (void)ws_size;
  const float* x        = (const float*)d_in[0];
  const int*   doc_lens = (const int*)d_in[1];
  const float* conv_w[3] = {(const float*)d_in[2], (const float*)d_in[4], (const float*)d_in[6]};
  const float* conv_b[3] = {(const float*)d_in[3], (const float*)d_in[5], (const float*)d_in[7]};
  const float* W_ih  = (const float*)d_in[8];
  const float* W_hh  = (const float*)d_in[9];
  const float* b_ih  = (const float*)d_in[10];
  const float* b_hh  = (const float*)d_in[11];
  const float* W_ihr = (const float*)d_in[12];
  const float* W_hhr = (const float*)d_in[13];
  const float* b_ihr = (const float*)d_in[14];
  const float* b_hhr = (const float*)d_in[15];
  float* out = (float*)d_out;

  const int N     = out_size / (2 * FHID);
  const int tiles = (N + 15) / 16;
  const int Npad  = tiles * 16;
  const int taps[3] = {3, 5, 7};

  // -------- workspace carve (fixed layout, 256B aligned)
  char* p = (char*)d_ws;
  auto carve = [&](size_t bytes) { char* r = p; p += (bytes + 255) & ~(size_t)255; return r; };
  int*   offs   = (int*)carve(sizeof(int) * 132);
  int*   posw   = (int*)carve(sizeof(int) * MAXN);
  u16*   xdoc0  = (u16*)carve(sizeof(u16) * (size_t)(MAXN + 48) * DPAD);
  u16*   xdoc   = xdoc0 + (size_t)8 * DPAD;                // 8 front / 40 back halo rows
  u16*   wpck[3];
  for (int j = 0; j < 3; ++j) wpck[j] = (u16*)carve(sizeof(u16) * (size_t)taps[j] * 256 * DPAD);
  u16*   ycat0  = (u16*)carve(sizeof(u16) * (size_t)(MAXN + 48) * FHID);
  u16*   ycat   = ycat0 + (size_t)8 * FHID;
  u16*   pooled = (u16*)carve(sizeof(u16) * (size_t)MAXN * FHID);
  u16*   Wihb   = (u16*)carve(sizeof(u16) * (size_t)G4 * FHID);
  u16*   Whhb   = (u16*)carve(sizeof(u16) * (size_t)G4 * FHID);
  u16*   Wihrb  = (u16*)carve(sizeof(u16) * (size_t)G4 * FHID);
  u16*   Whhrb  = (u16*)carve(sizeof(u16) * (size_t)G4 * FHID);
  float* bsumF  = (float*)carve(sizeof(float) * G4);
  float* bsumR  = (float*)carve(sizeof(float) * G4);
  float* gates  = (float*)carve(sizeof(float) * (size_t)MAXN * G4);
  u16*   h_all  = (u16*)carve(sizeof(u16) * (size_t)MAXN * FHID);
  float* c_all  = (float*)carve(sizeof(float) * (size_t)MAXN * FHID);

  auto blks = [](long total, int bs) { return (int)((total + bs - 1) / bs); };

  // -------- 1. doc offsets + in-doc positions
  k_offsets<<<1, 32, 0, stream>>>(doc_lens, offs);
  k_posw<<<128, 256, 0, stream>>>(offs, posw);

  // -------- 2. embed (x + PE) -> bf16, plus zero halos for conv
  k_embed<<<blks((long)N * DPAD, 256), 256, 0, stream>>>(x, posw, xdoc, N);
  k_fill_u16<<<blks((long)8 * DPAD, 256), 256, 0, stream>>>(xdoc - (size_t)8 * DPAD, 0, (long)8 * DPAD);
  k_fill_u16<<<blks((long)24 * DPAD, 256), 256, 0, stream>>>(xdoc + (size_t)N * DPAD, 0, (long)24 * DPAD);

  // -------- 3. weight packing (f32 -> bf16)
  k_pack_bf<<<blks((long)G4 * FHID, 256), 256, 0, stream>>>(W_ih,  Wihb,  (long)G4 * FHID);
  k_pack_bf<<<blks((long)G4 * FHID, 256), 256, 0, stream>>>(W_hh,  Whhb,  (long)G4 * FHID);
  k_pack_bf<<<blks((long)G4 * FHID, 256), 256, 0, stream>>>(W_ihr, Wihrb, (long)G4 * FHID);
  k_pack_bf<<<blks((long)G4 * FHID, 256), 256, 0, stream>>>(W_hhr, Whhrb, (long)G4 * FHID);
  k_pack_bias<<<blks(G4, 256), 256, 0, stream>>>(b_ih,  b_hh,  bsumF, G4);
  k_pack_bias<<<blks(G4, 256), 256, 0, stream>>>(b_ihr, b_hhr, bsumR, G4);
  for (int j = 0; j < 3; ++j)
    k_pack_conv<<<blks((long)taps[j] * 256 * DPAD, 256), 256, 0, stream>>>(conv_w[j], wpck[j], taps[j]);

  // -------- 4. convs (implicit GEMM, WMMA) + LeakyReLU, then -inf halo + maxpool
  for (int j = 0; j < 3; ++j)
    k_conv<<<dim3(tiles, 1, 1), 256, 0, stream>>>(xdoc, wpck[j], conv_b[j],
                                                  ycat + (size_t)j * 256, taps[j], taps[j] / 2);
  k_fill_u16<<<blks((long)8 * FHID, 256), 256, 0, stream>>>(ycat - (size_t)8 * FHID, 0xFF80u, (long)8 * FHID);
  k_fill_u16<<<blks((long)24 * FHID, 256), 256, 0, stream>>>(ycat + (size_t)N * FHID, 0xFF80u, (long)24 * FHID);
  k_pool<<<blks((long)Npad * FHID, 256), 256, 0, stream>>>(ycat, pooled, N, Npad);

  // -------- 5. x-side gate pre-GEMM for the scan: pooled @ W_ih^T + b_ih + b_hh
  k_gemm_gates<<<dim3(tiles, G4 / 256, 1), 256, 0, stream>>>(
      pooled, nullptr, 0, Wihb, Whhb, bsumF, gates, N);

  // -------- 6. persistent LSTM scan (1 WGP, 264KB LDS)
  k_scan<<<1, 1024, 16 * FHID * 2 + 16 * FHID * 4 + 16 * G4 * 4, stream>>>(
      gates, Whhb, h_all, c_all, N, tiles);

  // -------- 7. final cells: fwd (h/c shifted -9), rev (h/c shifted +9, rev weights)
  k_gemm_gates<<<dim3(tiles, G4 / 256, 1), 256, 0, stream>>>(
      ycat, h_all, -SHIFT, Wihb, Whhb, bsumF, gates, N);
  k_cellact<<<blks((long)N * FHID, 256), 256, 0, stream>>>(gates, c_all, -SHIFT, out, 0, N);

  k_gemm_gates<<<dim3(tiles, G4 / 256, 1), 256, 0, stream>>>(
      ycat, h_all, +SHIFT, Wihrb, Whhrb, bsumR, gates, N);
  k_cellact<<<blks((long)N * FHID, 256), 256, 0, stream>>>(gates, c_all, +SHIFT, out, FHID, N);
}
